// SimCLRLoss_8693013807618
// MI455X (gfx1250) — compile-verified, matrix operand order checked by reading, not run on hardware
//
#include <hip/hip_runtime.h>

// ---------------------------------------------------------------------------
// SimCLR (NT-Xent) loss, fused flash-softmax over v_wmma_f32_16x16x32_bf16.
//   N = 16384 rows (orig ++ aug), D = 128, T = 0.07
//   loss = mean_i( logsumexp_{j!=i}(f_i.f_j / T) - f_i.f_{(i+B)%N} / T )
//
// Pipeline:
//   1) cvt_kernel:    fp32 -> bf16 features, pre-scaled by sqrt(1/(T*ln2)) so
//                     WMMA dot products land directly in the log2 domain
//   2) simclr_kernel: per 16-row wave tile, loop over 1024 column tiles with
//                     WMMA bf16 (K=128) + online log2-domain softmax. The two
//                     special tiles (diagonal mask, positive capture) are
//                     PEELED out of the loop so the hot path has no
//                     divergence and no per-iteration compares.
//   3) reduce_kernel: mean of 16384 per-row losses -> d_out[0]
// ---------------------------------------------------------------------------

typedef __bf16 bf16_t;
typedef bf16_t v16bf __attribute__((ext_vector_type(16)));
typedef float  v8f   __attribute__((ext_vector_type(8)));

#define N_TOTAL 16384
#define DIM     128
#define B_HALF  8192
#define NTILES  (N_TOTAL / 16)

// sqrt(1/(0.07*ln2)): applied to both operands at conversion time, so
// dot(f_i, f_j) / (0.07*ln2) comes straight out of the accumulator.
#define PRESCALE 4.5398154f
#define LN2      0.69314718055994531f
#define MASK_NEG (-3.0e38f)   // finite "-inf": exp2(MASK_NEG - m) flushes to 0

// ---------------------------------------------------------------------------
// fp32 -> prescaled bf16 (round to nearest even), 4 elements per thread
// ---------------------------------------------------------------------------
__device__ __forceinline__ unsigned short f32_to_bf16(float f) {
    unsigned int u = __float_as_uint(f);
    u += 0x7FFFu + ((u >> 16) & 1u);
    return (unsigned short)(u >> 16);
}

__global__ __launch_bounds__(256) void cvt_kernel(
    const float* __restrict__ orig, const float* __restrict__ aug,
    unsigned short* __restrict__ F)
{
    const int t    = blockIdx.x * 256 + threadIdx.x;
    const int base = t * 4;                      // element index into [N*D]
    const int half = B_HALF * DIM;               // divisible by 4: no straddle
    const float* src = (base < half) ? (orig + base) : (aug + (base - half));
    float4 f = *(const float4*)src;
    ushort4 o;
    o.x = f32_to_bf16(f.x * PRESCALE);
    o.y = f32_to_bf16(f.y * PRESCALE);
    o.z = f32_to_bf16(f.z * PRESCALE);
    o.w = f32_to_bf16(f.w * PRESCALE);
    *(ushort4*)(F + base) = o;
}

// ---------------------------------------------------------------------------
// Fragment loader: 16-bit A/B layout for 16x16x32 WMMA.
//   lane L holds row (tileBase + L%16); lanes 0-15 carry K {kc+0..7,kc+16..23},
//   lanes 16-31 carry K {kc+8..15, kc+24..31} -> two 16-byte loads per frag.
// A (M-in-lanes) and B (N-in-lanes) patterns coincide for F * F^T.
// ---------------------------------------------------------------------------
union FragU {
    struct { uint4 lo; uint4 hi; } u;
    v16bf v;
};

__device__ __forceinline__ v16bf load_frag(const unsigned short* __restrict__ F,
                                           int row, int kc, int lane)
{
    const int koff = (lane >> 4) << 3;           // 0 or 8 bf16 elements
    const unsigned short* p = F + row * DIM + kc + koff;
    FragU f;
    f.u.lo = *(const uint4*)(p);
    f.u.hi = *(const uint4*)(p + 16);
    return f.v;
}

// One 16x16 logit tile: 8 b128 loads + prefetch + 4 chained WMMAs (K=128).
__device__ __forceinline__ v8f mma_tile(const unsigned short* __restrict__ F,
                                        int ct, int nIdx, int lane,
                                        v16bf a0, v16bf a1, v16bf a2, v16bf a3)
{
    const int bRow = (ct << 4) + nIdx;
    const v16bf b0 = load_frag(F, bRow,  0, lane);
    const v16bf b1 = load_frag(F, bRow, 32, lane);
    const v16bf b2 = load_frag(F, bRow, 64, lane);
    const v16bf b3 = load_frag(F, bRow, 96, lane);
    // Prefetch next column tile (global_prefetch_b8); speculative, the one
    // off-the-end prefetch lands in workspace and is harmless.
    __builtin_prefetch(F + (bRow + 16) * DIM + ((lane >> 4) ? 64 : 0), 0, 3);
    v8f c = {};
    c = __builtin_amdgcn_wmma_f32_16x16x32_bf16(false, a0, false, b0, (short)0, c, false, false);
    c = __builtin_amdgcn_wmma_f32_16x16x32_bf16(false, a1, false, b1, (short)0, c, false, false);
    c = __builtin_amdgcn_wmma_f32_16x16x32_bf16(false, a2, false, b2, (short)0, c, false, false);
    c = __builtin_amdgcn_wmma_f32_16x16x32_bf16(false, a3, false, b3, (short)0, c, false, false);
    return c;
}

// Hot-path online softmax update (log2 domain): no masks, no compares.
__device__ __forceinline__ void upd_plain(const v8f& c, float m[8], float s[8])
{
#pragma unroll
    for (int r = 0; r < 8; ++r) {
        const float v  = c[r];
        const float mN = fmaxf(m[r], v);
        s[r] = s[r] * __builtin_amdgcn_exp2f(m[r] - mN)
             +        __builtin_amdgcn_exp2f(v    - mN);
        m[r] = mN;
    }
}

// Cold path (executed twice per wave): branchless cndmask selects for the
// diagonal mask and the positive-logit capture.
__device__ __forceinline__ void upd_special(const v8f& c, float m[8], float s[8],
                                            float pos[8], bool isDiag, bool isPos,
                                            int nIdx, int mHi)
{
#pragma unroll
    for (int r = 0; r < 8; ++r) {
        float v = c[r];
        const bool match = (nIdx == r + mHi);    // j - i == const line of the tile
        if (isPos)  pos[r] = match ? v : pos[r];
        if (isDiag) v      = match ? MASK_NEG : v;
        const float mN = fmaxf(m[r], v);
        s[r] = s[r] * __builtin_amdgcn_exp2f(m[r] - mN)
             +        __builtin_amdgcn_exp2f(v    - mN);
        m[r] = mN;
    }
}

// ---------------------------------------------------------------------------
// Main fused kernel: 256 threads = 8 waves; wave w owns rows
// [blockIdx.x*128 + w*16, +16). Grid = N/128 = 128 blocks.
// ---------------------------------------------------------------------------
__global__ __launch_bounds__(256) void simclr_kernel(
    const unsigned short* __restrict__ F, float* __restrict__ rowloss)
{
    const int lane    = threadIdx.x & 31;
    const int wave    = threadIdx.x >> 5;
    const int rowBase = blockIdx.x * 128 + wave * 16;
    const int nIdx    = lane & 15;
    const int mHi     = (lane >> 4) << 3;        // 0 for lanes 0-15, 8 for 16-31
    const int aRow    = rowBase + nIdx;

    // Resident A fragments covering the full K = 128 (32 VGPRs).
    const v16bf a0 = load_frag(F, aRow,  0, lane);
    const v16bf a1 = load_frag(F, aRow, 32, lane);
    const v16bf a2 = load_frag(F, aRow, 64, lane);
    const v16bf a3 = load_frag(F, aRow, 96, lane);

    // Online softmax state per accumulator slot (8 regs x 32 lanes = 16x16
    // elements; slot r in lane L is row M = r + mHi, col N = L%16).
    float m[8], s[8], pos[8];
#pragma unroll
    for (int r = 0; r < 8; ++r) { m[r] = -INFINITY; s[r] = 0.0f; pos[r] = 0.0f; }

    const int diagTile = rowBase >> 4;                               // j == i
    const int posTile  = ((rowBase + B_HALF) & (N_TOTAL - 1)) >> 4;  // j == (i+B)%N
    const int t0 = (diagTile < posTile) ? diagTile : posTile;
    const int t1 = (diagTile < posTile) ? posTile  : diagTile;       // t0 < t1 always

    // --- three clean ranges + two peeled special tiles ---------------------
    for (int ct = 0; ct < t0; ++ct) {
        const v8f c = mma_tile(F, ct, nIdx, lane, a0, a1, a2, a3);
        upd_plain(c, m, s);
    }
    {
        const v8f c = mma_tile(F, t0, nIdx, lane, a0, a1, a2, a3);
        upd_special(c, m, s, pos, t0 == diagTile, t0 == posTile, nIdx, mHi);
    }
    for (int ct = t0 + 1; ct < t1; ++ct) {
        const v8f c = mma_tile(F, ct, nIdx, lane, a0, a1, a2, a3);
        upd_plain(c, m, s);
    }
    {
        const v8f c = mma_tile(F, t1, nIdx, lane, a0, a1, a2, a3);
        upd_special(c, m, s, pos, t1 == diagTile, t1 == posTile, nIdx, mHi);
    }
    for (int ct = t1 + 1; ct < NTILES; ++ct) {
        const v8f c = mma_tile(F, ct, nIdx, lane, a0, a1, a2, a3);
        upd_plain(c, m, s);
    }

    // Merge (m, s, pos) across the 16 lanes of each half-wave (butterfly).
#pragma unroll
    for (int r = 0; r < 8; ++r) {
        float mm = m[r], ss = s[r], pp = pos[r];
#pragma unroll
        for (int off = 1; off < 16; off <<= 1) {
            const float mo = __shfl_xor(mm, off, 32);
            const float so = __shfl_xor(ss, off, 32);
            const float po = __shfl_xor(pp, off, 32);
            const float mN = fmaxf(mm, mo);
            ss = ss * __builtin_amdgcn_exp2f(mm - mN)
               + so * __builtin_amdgcn_exp2f(mo - mN);
            mm = mN;
            pp += po;                            // exactly one lane held pos
        }
        if (nIdx == 0) {
            const int row = rowBase + r + mHi;
            // back to natural log: loss_i = ln2 * (m + log2(s) - pos_log2)
            rowloss[row] = LN2 * (mm + __builtin_amdgcn_logf(ss) - pp);
        }
    }
}

// ---------------------------------------------------------------------------
// Mean over 16384 row losses -> scalar.
// ---------------------------------------------------------------------------
__global__ __launch_bounds__(256) void reduce_kernel(
    const float* __restrict__ rowloss, float* __restrict__ out)
{
    __shared__ float sm[256];
    float acc = 0.0f;
    for (int i = threadIdx.x; i < N_TOTAL; i += 256) acc += rowloss[i];
    sm[threadIdx.x] = acc;
    __syncthreads();
    for (int off = 128; off > 0; off >>= 1) {
        if ((int)threadIdx.x < off) sm[threadIdx.x] += sm[threadIdx.x + off];
        __syncthreads();
    }
    if (threadIdx.x == 0) out[0] = sm[0] * (1.0f / (float)N_TOTAL);
}

// ---------------------------------------------------------------------------
extern "C" void kernel_launch(void* const* d_in, const int* in_sizes, int n_in,
                              void* d_out, int out_size, void* d_ws, size_t ws_size,
                              hipStream_t stream)
{
    (void)in_sizes; (void)n_in; (void)out_size; (void)ws_size;

    const float* orig = (const float*)d_in[0];   // [8192, 128] fp32
    const float* aug  = (const float*)d_in[1];   // [8192, 128] fp32

    unsigned short* Fbf    = (unsigned short*)d_ws;                       // 4 MB
    float*          rowlss = (float*)((char*)d_ws +
                              (size_t)N_TOTAL * DIM * sizeof(unsigned short)); // 64 KB

    const int total = N_TOTAL * DIM;             // 2,097,152 elements
    cvt_kernel<<<total / (256 * 4), 256, 0, stream>>>(orig, aug, Fbf);
    simclr_kernel<<<N_TOTAL / 128, 256, 0, stream>>>(Fbf, rowlss);
    reduce_kernel<<<1, 256, 0, stream>>>(rowlss, (float*)d_out);
}
